// SwinTransformerBlock_56143812494070
// MI455X (gfx1250) — matrix operand order repaired and checked by reading, not compiled, not run on hardware
//
#include <hip/hip_runtime.h>
#include <math.h>

// ---------------- problem constants ----------------
#define CC     256
#define HEADS  8
#define WSZ    8
#define SHIFT  4
#define HD     32
#define NTOK   64      // tokens per window
#define NWTOT  4096    // B * nW
#define SCALE  0.17677669529663689f  // 1/sqrt(32)

typedef __attribute__((ext_vector_type(16))) __bf16 v16bf;
typedef __attribute__((ext_vector_type(8)))  float  v8f;

union FragAB { v16bf v; uint4 q[2]; };

// native f32 -> bf16 (lowered to v_cvt_*bf16_f32, RNE)
__device__ __forceinline__ unsigned short f2bf(float f) {
    union { __bf16 b; unsigned short u; } c;
    c.b = (__bf16)f;
    return c.u;
}
// packed pair (lowered to v_cvt_pk_bf16_f32 when available)
__device__ __forceinline__ unsigned pack2bf(float lo, float hi) {
    union { __bf16 b[2]; unsigned u; } c;
    c.b[0] = (__bf16)lo; c.b[1] = (__bf16)hi;
    return c.u;
}

// A fragment (16x32 bf16): lane = row (l&15); K chunks {kb..kb+7, kb+16..kb+23}, kb = (l>=16)?8:0
__device__ __forceinline__ v16bf ldA(const void* base, int strideBytes, int lane) {
    int r  = lane & 15;
    int kb = (lane & 16) ? 16 : 0;                     // byte offset of kb elems
    const char* p = (const char*)base + r * strideBytes + kb;
    FragAB f;
    f.q[0] = *(const uint4*)(p);
    f.q[1] = *(const uint4*)(p + 32);
    return f.v;
}

// B fragment (32x16 bf16), source is row-major [n][k] (k contiguous):
// lane = col n (l&15); K chunks {kb..kb+7, kb+16..kb+23}, kb = (l>=16)?8:0
__device__ __forceinline__ v16bf ldB(const void* base, int strideBytes, int lane) {
    int n  = lane & 15;
    int kb = (lane & 16) ? 16 : 0;
    const char* p = (const char*)base + n * strideBytes + kb;
    FragAB f;
    f.q[0] = *(const uint4*)(p);
    f.q[1] = *(const uint4*)(p + 32);
    return f.v;
}

__device__ __forceinline__ v8f wmma_bf16(v16bf a, v16bf b, v8f c) {
    return __builtin_amdgcn_wmma_f32_16x16x32_bf16(false, a, false, b, (short)0, c, false, false);
}

// ---------------- LDS layouts (bytes) ----------------
#define ATT_ROWOFF 0
#define ATT_REG    256
#define ATT_XL     512                       // 64 x 264 bf16 (row 528B)
#define ATT_Q      (ATT_XL + 64*528)         // 8 x 64 x 40 bf16 (row 80B)
#define ATT_K      (ATT_Q  + 8*64*80)
#define ATT_VT     (ATT_K  + 8*64*80)        // 8 x 32 x 72 bf16 (row 144B), V transposed
#define ATT_O      (ATT_VT + 8*32*144)       // 64 x 264 bf16
#define ATT_P      (ATT_O  + 64*528)         // per-wave 64 x 72 bf16
#define ATT_SMEM   (ATT_P  + 8*64*144)       // 260608 B

#define MLP_XM     0                         // 64 x 264 bf16
#define MLP_H1     (64*528)                  // 64 x 1032 bf16 (row 2064B)
#define MLP_SMEM   (MLP_H1 + 64*2064)        // 165888 B

// ---------------- workspace layout (bytes) ----------------
#define WS_WQKV  0u
#define WS_WPROJ 393216u
#define WS_W1    524288u
#define WS_W2    1048576u
#define WS_X2    1572864u                    // 67,108,864 f32

__global__ void k_f32_to_bf16(const float* __restrict__ s, unsigned short* __restrict__ d, int n) {
    int i = blockIdx.x * blockDim.x + threadIdx.x;
    if (i < n) d[i] = f2bf(s[i]);
}

// ================= fused window-attention block =================
// One workgroup (8 waves) per window: LN1 + shifted-window gather -> QKV GEMM ->
// per-head (wave==head) scores + rel-bias + shift-mask + softmax + PV -> proj + residual.
__global__ void __launch_bounds__(256, 1) k_attn(
    const float* __restrict__ x,
    const float* __restrict__ n1w, const float* __restrict__ n1b,
    const unsigned short* __restrict__ wqkv, const float* __restrict__ bqkv,
    const float* __restrict__ relt,
    const unsigned short* __restrict__ wproj, const float* __restrict__ bproj,
    float* __restrict__ x2)
{
    extern __shared__ char sm[];
    const int tid = threadIdx.x, lane = tid & 31, wv = tid >> 5;
    const int hi = (lane >> 4) & 1, ln = lane & 15;
    const int wi = blockIdx.x;
    const int b = wi >> 10, wr = wi & 1023, wh = wr >> 5, wwc = wr & 31;

    int* rowoff = (int*)(sm + ATT_ROWOFF);
    int* regid  = (int*)(sm + ATT_REG);
    if (tid < 64) {
        int th = tid >> 3, tw = tid & 7;
        int rh = wh * 8 + th, rw = wwc * 8 + tw;              // rolled coords
        int oh = (rh + SHIFT) & 255, ow = (rw + SHIFT) & 255; // original coords
        rowoff[tid] = b * 65536 + oh * 256 + ow;
        int fh = (rh < 248) ? 0 : ((rh < 252) ? 1 : 2);
        int fw = (rw < 248) ? 0 : ((rw < 252) ? 1 : 2);
        regid[tid] = fh * 3 + fw;
    }
    __syncthreads();

    // ---- LN1 + gather to LDS (bf16) ----
    {
        int t = tid >> 2, q = tid & 3;
        const float* xr = x + (size_t)rowoff[t] * CC + q * 64;
        float s = 0.f, sq = 0.f;
        for (int c = 0; c < 64; c += 4) {
            float4 v4 = *(const float4*)(xr + c);
            s  += v4.x + v4.y + v4.z + v4.w;
            sq += v4.x*v4.x + v4.y*v4.y + v4.z*v4.z + v4.w*v4.w;
        }
        s  += __shfl_xor(s, 1);  s  += __shfl_xor(s, 2);
        sq += __shfl_xor(sq, 1); sq += __shfl_xor(sq, 2);
        float mu  = s * (1.f / 256.f);
        float var = sq * (1.f / 256.f) - mu * mu;
        float rs  = rsqrtf(var + 1e-5f);
        unsigned short* xl = (unsigned short*)(sm + ATT_XL) + t * 264 + q * 64;
        for (int c = 0; c < 64; c += 2) {
            int ch = q * 64 + c;
            float v0 = (xr[c]   - mu) * rs * n1w[ch]   + n1b[ch];
            float v1 = (xr[c+1] - mu) * rs * n1w[ch+1] + n1b[ch+1];
            *(unsigned*)(xl + c) = pack2bf(v0, v1);
        }
    }
    __syncthreads();

    // ---- QKV GEMM: [64x256] @ [256x768] ; wave handles 6 n-tiles x 4 m-tiles ----
    for (int j = 0; j < 6; j++) {
        int nt = wv * 6 + j;
        v8f acc[4] = {};
        for (int ks = 0; ks < 8; ks++) {
            if (ks < 7)  // prefetch next k-step weight row (global_prefetch_b8)
                __builtin_prefetch(wqkv + nt * 16 * 256 + (ks + 1) * 32 + (lane & 15) * 256, 0, 1);
            v16bf bf = ldB(wqkv + nt * 16 * 256 + ks * 32, 512, lane);
            #pragma unroll
            for (int mt = 0; mt < 4; mt++) {
                v16bf af = ldA(sm + ATT_XL + mt * 16 * 528 + ks * 64, 528, lane);
                acc[mt] = wmma_bf16(af, bf, acc[mt]);
            }
        }
        int col = nt * 16 + ln;
        float bias = bqkv[col];
        #pragma unroll
        for (int mt = 0; mt < 4; mt++) {
            #pragma unroll
            for (int e = 0; e < 8; e++) {
                float v = acc[mt][e] + bias;
                int tok = mt * 16 + e + hi * 8;
                if (col < 256) {
                    int hd = col >> 5, d = col & 31;
                    *(unsigned short*)(sm + ATT_Q + ((hd * 64 + tok) * 40 + d) * 2) = f2bf(v * SCALE);
                } else if (col < 512) {
                    int c2 = col - 256, hd = c2 >> 5, d = c2 & 31;
                    *(unsigned short*)(sm + ATT_K + ((hd * 64 + tok) * 40 + d) * 2) = f2bf(v);
                } else {
                    int c2 = col - 512, hd = c2 >> 5, d = c2 & 31;
                    *(unsigned short*)(sm + ATT_VT + ((hd * 32 + d) * 72 + tok) * 2) = f2bf(v);
                }
            }
        }
    }
    __syncthreads();

    // ---- attention: wave == head ----
    {
        const int h = wv;
        v8f s[4][4] = {};
        #pragma unroll
        for (int nt = 0; nt < 4; nt++) {
            v16bf bk = ldB(sm + ATT_K + (h * 64 + nt * 16) * 80, 80, lane);
            #pragma unroll
            for (int mt = 0; mt < 4; mt++) {
                v16bf aq = ldA(sm + ATT_Q + (h * 64 + mt * 16) * 80, 80, lane);
                s[mt][nt] = wmma_bf16(aq, bk, s[mt][nt]);       // K dim = HD = 32: one WMMA
            }
        }
        // relative-position bias + shift mask
        #pragma unroll
        for (int nt = 0; nt < 4; nt++) {
            int nTok = nt * 16 + ln;           // key token
            int rn = regid[nTok];
            int nhh = nTok >> 3, nww = nTok & 7;
            #pragma unroll
            for (int mt = 0; mt < 4; mt++) {
                #pragma unroll
                for (int e = 0; e < 8; e++) {
                    int mTok = mt * 16 + e + hi * 8;  // query token
                    int rid = ((mTok >> 3) - nhh + 7) * 15 + ((mTok & 7) - nww + 7);
                    float add = relt[rid * 8 + h] + ((regid[mTok] == rn) ? 0.f : -100.f);
                    s[mt][nt][e] = s[mt][nt][e] + add;
                }
            }
        }
        // row softmax via cross-lane reductions within 16-lane halves
        #pragma unroll
        for (int mt = 0; mt < 4; mt++) {
            #pragma unroll
            for (int e = 0; e < 8; e++) {
                float mx = fmaxf(fmaxf(s[mt][0][e], s[mt][1][e]), fmaxf(s[mt][2][e], s[mt][3][e]));
                mx = fmaxf(mx, __shfl_xor(mx, 1));
                mx = fmaxf(mx, __shfl_xor(mx, 2));
                mx = fmaxf(mx, __shfl_xor(mx, 4));
                mx = fmaxf(mx, __shfl_xor(mx, 8));
                float sum = 0.f;
                #pragma unroll
                for (int nt = 0; nt < 4; nt++) {
                    float ev = __expf(s[mt][nt][e] - mx);      // hardware v_exp_f32
                    s[mt][nt][e] = ev; sum += ev;
                }
                sum += __shfl_xor(sum, 1); sum += __shfl_xor(sum, 2);
                sum += __shfl_xor(sum, 4); sum += __shfl_xor(sum, 8);
                float inv = 1.f / sum;
                #pragma unroll
                for (int nt = 0; nt < 4; nt++) s[mt][nt][e] = s[mt][nt][e] * inv;
            }
        }
        // write P (wave-private; DS is in-order within a wave, no barrier needed)
        char* pb = sm + ATT_P + wv * 64 * 144;
        #pragma unroll
        for (int mt = 0; mt < 4; mt++)
            #pragma unroll
            for (int nt = 0; nt < 4; nt++)
                #pragma unroll
                for (int e = 0; e < 8; e++) {
                    int mTok = mt * 16 + e + hi * 8;
                    *(unsigned short*)(pb + (mTok * 72 + nt * 16 + ln) * 2) = f2bf(s[mt][nt][e]);
                }
        // O = P @ V  (B fragments from transposed V)
        v8f o[4][2] = {};
        #pragma unroll
        for (int ks = 0; ks < 2; ks++) {
            #pragma unroll
            for (int dt = 0; dt < 2; dt++) {
                v16bf bv = ldB(sm + ATT_VT + (h * 32 + dt * 16) * 144 + ks * 64, 144, lane);
                #pragma unroll
                for (int mt = 0; mt < 4; mt++) {
                    v16bf ap = ldA(pb + mt * 16 * 144 + ks * 64, 144, lane);
                    o[mt][dt] = wmma_bf16(ap, bv, o[mt][dt]);
                }
            }
        }
        #pragma unroll
        for (int mt = 0; mt < 4; mt++)
            #pragma unroll
            for (int dt = 0; dt < 2; dt++)
                #pragma unroll
                for (int e = 0; e < 8; e++) {
                    int tok = mt * 16 + e + hi * 8;
                    int col = h * 32 + dt * 16 + ln;
                    *(unsigned short*)(sm + ATT_O + (tok * 264 + col) * 2) = f2bf(o[mt][dt][e]);
                }
    }
    __syncthreads();

    // ---- proj + shortcut residual -> x2 (un-partition + un-roll via rowoff) ----
    {
        int mt = wv >> 1;
        int ntB = (wv & 1) * 8;
        v8f pacc[8] = {};
        for (int ks = 0; ks < 8; ks++) {
            v16bf a = ldA(sm + ATT_O + mt * 16 * 528 + ks * 64, 528, lane);
            #pragma unroll
            for (int j = 0; j < 8; j++) {
                v16bf bW = ldB(wproj + (ntB + j) * 16 * 256 + ks * 32, 512, lane);
                pacc[j] = wmma_bf16(a, bW, pacc[j]);
            }
        }
        #pragma unroll
        for (int j = 0; j < 8; j++) {
            int col = (ntB + j) * 16 + ln;
            float pb2 = bproj[col];
            #pragma unroll
            for (int e = 0; e < 8; e++) {
                int tok = mt * 16 + e + hi * 8;
                size_t off = (size_t)rowoff[tok] * CC + col;
                x2[off] = x[off] + pacc[j][e] + pb2;
            }
        }
    }
}

// ================= fused MLP block =================
// One workgroup per 64 token rows: LN2 -> FC1+exact GELU (64x1024 tile in LDS) -> FC2 + residual.
__global__ void __launch_bounds__(256, 1) k_mlp(
    const float* __restrict__ x2,
    const float* __restrict__ n2w, const float* __restrict__ n2b,
    const unsigned short* __restrict__ w1, const float* __restrict__ b1,
    const unsigned short* __restrict__ w2, const float* __restrict__ b2,
    float* __restrict__ out)
{
    extern __shared__ char sm[];
    const int tid = threadIdx.x, lane = tid & 31, wv = tid >> 5;
    const int hi = (lane >> 4) & 1, ln = lane & 15;
    const size_t rowbase = (size_t)blockIdx.x * 64;

    // ---- LN2 ----
    {
        int t = tid >> 2, q = tid & 3;
        const float* xr = x2 + (rowbase + t) * CC + q * 64;
        float s = 0.f, sq = 0.f;
        for (int c = 0; c < 64; c += 4) {
            float4 v4 = *(const float4*)(xr + c);
            s  += v4.x + v4.y + v4.z + v4.w;
            sq += v4.x*v4.x + v4.y*v4.y + v4.z*v4.z + v4.w*v4.w;
        }
        s  += __shfl_xor(s, 1);  s  += __shfl_xor(s, 2);
        sq += __shfl_xor(sq, 1); sq += __shfl_xor(sq, 2);
        float mu  = s * (1.f / 256.f);
        float var = sq * (1.f / 256.f) - mu * mu;
        float rs  = rsqrtf(var + 1e-5f);
        unsigned short* xm = (unsigned short*)(sm + MLP_XM) + t * 264 + q * 64;
        for (int c = 0; c < 64; c += 2) {
            int ch = q * 64 + c;
            float v0 = (xr[c]   - mu) * rs * n2w[ch]   + n2b[ch];
            float v1 = (xr[c+1] - mu) * rs * n2w[ch+1] + n2b[ch+1];
            *(unsigned*)(xm + c) = pack2bf(v0, v1);
        }
    }
    __syncthreads();

    // ---- FC1 + exact GELU -> h1 in LDS ----
    for (int j = 0; j < 8; j++) {
        int nt = wv * 8 + j;
        v8f acc[4] = {};
        for (int ks = 0; ks < 8; ks++) {
            if (ks < 7)
                __builtin_prefetch(w1 + nt * 16 * 256 + (ks + 1) * 32 + (lane & 15) * 256, 0, 1);
            v16bf bW = ldB(w1 + nt * 16 * 256 + ks * 32, 512, lane);
            #pragma unroll
            for (int mt = 0; mt < 4; mt++) {
                v16bf a = ldA(sm + MLP_XM + mt * 16 * 528 + ks * 64, 528, lane);
                acc[mt] = wmma_bf16(a, bW, acc[mt]);
            }
        }
        int col = nt * 16 + ln;
        float bb = b1[col];
        #pragma unroll
        for (int mt = 0; mt < 4; mt++)
            #pragma unroll
            for (int e = 0; e < 8; e++) {
                float v = acc[mt][e] + bb;
                float g = 0.5f * v * (1.f + erff(v * 0.70710678118654752f));
                int tok = mt * 16 + e + hi * 8;
                *(unsigned short*)(sm + MLP_H1 + (tok * 1032 + col) * 2) = f2bf(g);
            }
    }
    __syncthreads();

    // ---- FC2 + residual -> out ----
    for (int j = 0; j < 2; j++) {
        int nt = wv * 2 + j;
        v8f acc[4] = {};
        for (int ks = 0; ks < 32; ks++) {
            if (ks < 31)
                __builtin_prefetch(w2 + nt * 16 * 1024 + (ks + 1) * 32 + (lane & 15) * 1024, 0, 1);
            v16bf bW = ldB(w2 + nt * 16 * 1024 + ks * 32, 2048, lane);
            #pragma unroll
            for (int mt = 0; mt < 4; mt++) {
                v16bf a = ldA(sm + MLP_H1 + mt * 16 * 2064 + ks * 64, 2064, lane);
                acc[mt] = wmma_bf16(a, bW, acc[mt]);
            }
        }
        int col = nt * 16 + ln;
        float bb = b2[col];
        #pragma unroll
        for (int mt = 0; mt < 4; mt++)
            #pragma unroll
            for (int e = 0; e < 8; e++) {
                int tok = mt * 16 + e + hi * 8;
                size_t off = (rowbase + tok) * CC + col;
                out[off] = x2[off] + acc[mt][e] + bb;
            }
    }
}

extern "C" void kernel_launch(void* const* d_in, const int* in_sizes, int n_in,
                              void* d_out, int out_size, void* d_ws, size_t ws_size,
                              hipStream_t stream)
{
    (void)in_sizes; (void)n_in; (void)out_size; (void)ws_size;
    const float* x     = (const float*)d_in[0];
    const float* n1w   = (const float*)d_in[1];
    const float* n1b   = (const float*)d_in[2];
    const float* qkv_w = (const float*)d_in[3];
    const float* qkv_b = (const float*)d_in[4];
    const float* relt  = (const float*)d_in[5];
    const float* pj_w  = (const float*)d_in[6];
    const float* pj_b  = (const float*)d_in[7];
    const float* n2w   = (const float*)d_in[8];
    const float* n2b   = (const float*)d_in[9];
    const float* fc1_w = (const float*)d_in[10];
    const float* fc1_b = (const float*)d_in[11];
    const float* fc2_w = (const float*)d_in[12];
    const float* fc2_b = (const float*)d_in[13];

    char* ws = (char*)d_ws;
    unsigned short* wqkv_bf  = (unsigned short*)(ws + WS_WQKV);
    unsigned short* wproj_bf = (unsigned short*)(ws + WS_WPROJ);
    unsigned short* w1_bf    = (unsigned short*)(ws + WS_W1);
    unsigned short* w2_bf    = (unsigned short*)(ws + WS_W2);
    float*          x2       = (float*)(ws + WS_X2);

    hipFuncSetAttribute((const void*)k_attn, hipFuncAttributeMaxDynamicSharedMemorySize, ATT_SMEM);
    hipFuncSetAttribute((const void*)k_mlp,  hipFuncAttributeMaxDynamicSharedMemorySize, MLP_SMEM);

    k_f32_to_bf16<<<(196608 + 255) / 256, 256, 0, stream>>>(qkv_w, wqkv_bf, 196608);
    k_f32_to_bf16<<<(65536  + 255) / 256, 256, 0, stream>>>(pj_w,  wproj_bf, 65536);
    k_f32_to_bf16<<<(262144 + 255) / 256, 256, 0, stream>>>(fc1_w, w1_bf,   262144);
    k_f32_to_bf16<<<(262144 + 255) / 256, 256, 0, stream>>>(fc2_w, w2_bf,   262144);

    k_attn<<<NWTOT, 256, ATT_SMEM, stream>>>(x, n1w, n1b, wqkv_bf, qkv_b, relt,
                                             wproj_bf, pj_b, x2);
    k_mlp<<<NWTOT, 256, MLP_SMEM, stream>>>(x2, n2w, n2b, w1_bf, fc1_b, w2_bf, fc2_b,
                                            (float*)d_out);
}
